// BiLSTM_CRF_89678917140835
// MI455X (gfx1250) — matrix operand order repaired and checked by reading, not compiled, not run on hardware
//
#include <hip/hip_runtime.h>
#include <hip/hip_bf16.h>
#include <math.h>

#define T_SEQ 512
#define BATCH 32
#define EMBD  256
#define HID4  1024
#define HDIM  256
#define KTAGS 9

typedef __bf16 bf16_t;
typedef __attribute__((ext_vector_type(16))) __bf16 v16bf;
typedef __attribute__((ext_vector_type(8)))  __bf16 v8bf;
typedef __attribute__((ext_vector_type(8)))  float  v8f;

static __device__ __forceinline__ v16bf cat16(v8bf lo, v8bf hi) {
  v16bf r;
#pragma unroll
  for (int i = 0; i < 8; ++i) { r[i] = lo[i]; r[i + 8] = hi[i]; }
  return r;
}

static __device__ __forceinline__ v8f wmma_bf16(v16bf a, v16bf b, v8f c) {
  // D = A(16x32) * B(32x16) + C, f32 accumulate
  return __builtin_amdgcn_wmma_f32_16x16x32_bf16(false, a, false, b, (short)0, c,
                                                 false, false);
}

// A fragment (16x32) from row-major bf16 matrix: lane m=lane&15,
// 16 contiguous K [h*16, h*16+16), h = lane>>4.
static __device__ __forceinline__ v16bf load_a_frag(const bf16_t* p, int ld,
                                                    int row0, int k0, int lane) {
  int m = lane & 15, h = lane >> 4;
  const bf16_t* base = p + (size_t)(row0 + m) * ld + k0 + h * 16;
  v8bf c0 = *(const v8bf*)(base);
  v8bf c1 = *(const v8bf*)(base + 8);
  return cat16(c0, c1);
}

// B fragment (32x16): column n of B = row n of row-major W (contiguous along K).
// lane n=lane&15 holds 16 contiguous K starting at k0 + 16*(lane>>4).
static __device__ __forceinline__ v16bf load_b_frag(const bf16_t* p, int ld,
                                                    int n0, int k0, int lane) {
  int n = lane & 15, h = lane >> 4;
  const bf16_t* base = p + (size_t)(n0 + n) * ld + k0 + h * 16;
  v8bf c0 = *(const v8bf*)(base);
  v8bf c1 = *(const v8bf*)(base + 8);
  return cat16(c0, c1);
}

static __device__ __forceinline__ float sigmoidf_(float x) {
  return 1.0f / (1.0f + __expf(-x));
}

// ---------------- fp32 -> bf16 weight conversion ----------------
__global__ __launch_bounds__(256) void cvt_kernel(const float* __restrict__ in,
                                                  bf16_t* __restrict__ out, int n) {
  int i = blockIdx.x * 256 + threadIdx.x;
  if (i < n) out[i] = (bf16_t)in[i];
}

// ---------------- embedding gather -> bf16 X (row = t*32+b) ----------------
__global__ __launch_bounds__(256) void embed_kernel(const int* __restrict__ texts,
                                                    const float* __restrict__ emb,
                                                    bf16_t* __restrict__ xbf) {
  int row = blockIdx.x;  // t*32 + b
  int e = threadIdx.x;
  int t = row >> 5, b = row & 31;
  int tok = texts[b * T_SEQ + t];
  xbf[(size_t)row * EMBD + e] = (bf16_t)emb[(size_t)tok * EMBD + e];
}

// -------- input projection GEMM: G[t][n][b] = (X @ Wih^T)[t*32+b][n] + bias[n] --------
__global__ __launch_bounds__(256) void ip_gemm_kernel(const bf16_t* __restrict__ X,
                                                      const bf16_t* __restrict__ W,
                                                      const float* __restrict__ bias,
                                                      float* __restrict__ G) {
  int lane = threadIdx.x & 31;
  int warp = threadIdx.x >> 5;
  int wj = blockIdx.x * 8 + warp;  // 0..16383 wave jobs
  int mst = wj >> 5;               // 0..511 (2 M-tiles each)
  int nst = wj & 31;               // 0..31  (2 N-tiles each)
  int m0 = mst * 32, n0 = nst * 32;
  v8f acc[2][2];
#pragma unroll
  for (int a = 0; a < 2; ++a)
#pragma unroll
    for (int b = 0; b < 2; ++b)
#pragma unroll
      for (int r = 0; r < 8; ++r) acc[a][b][r] = 0.0f;
#pragma unroll
  for (int kt = 0; kt < 8; ++kt) {
    int k0 = kt * 32;
    v16bf a0 = load_a_frag(X, EMBD, m0, k0, lane);
    v16bf a1 = load_a_frag(X, EMBD, m0 + 16, k0, lane);
    v16bf b0 = load_b_frag(W, EMBD, n0, k0, lane);
    v16bf b1 = load_b_frag(W, EMBD, n0 + 16, k0, lane);
    acc[0][0] = wmma_bf16(a0, b0, acc[0][0]);
    acc[0][1] = wmma_bf16(a0, b1, acc[0][1]);
    acc[1][0] = wmma_bf16(a1, b0, acc[1][0]);
    acc[1][1] = wmma_bf16(a1, b1, acc[1][1]);
  }
#pragma unroll
  for (int mi = 0; mi < 2; ++mi) {
#pragma unroll
    for (int ni = 0; ni < 2; ++ni) {
      int n = n0 + ni * 16 + (lane & 15);
      float bv = bias[n];
#pragma unroll
      for (int r = 0; r < 8; ++r) {
        int m = m0 + mi * 16 + r + 8 * (lane >> 4);
        int t = m >> 5, b = m & 31;
        G[((size_t)t * HID4 + n) * BATCH + b] = acc[mi][ni][r] + bv;
      }
    }
  }
}

// -------- sequential bi-LSTM: one workgroup per direction, 16 waves --------
// Recurrent GEMM computed transposed: A = Whh rows (gates), B = h^T (batch cols).
// Wave `warp` owns hidden units [warp*16, warp*16+16): gate rows j0, j0+256,
// j0+512, j0+768 -> all four gates for the same (j,b) land in matching
// (lane,reg) slots, so the cell update is element-wise in registers; c never
// leaves VGPRs. h is double-buffered in LDS (bf16) -> ONE barrier per step.
// Next step's gate pre-activations are prefetched to overlap the WMMA phase.
__global__ __launch_bounds__(512) void lstm_kernel(const float* __restrict__ Gf,
                                                   const float* __restrict__ Gr,
                                                   const bf16_t* __restrict__ WhhF,
                                                   const bf16_t* __restrict__ WhhR,
                                                   const float* __restrict__ h0,
                                                   const float* __restrict__ c0,
                                                   float* __restrict__ hf,
                                                   float* __restrict__ hr) {
  const int dir = blockIdx.x;
  const float* Gd = dir ? Gr : Gf;
  const bf16_t* Whh = dir ? WhhR : WhhF;
  float* hout = dir ? hr : hf;
  __shared__ bf16_t h_sh[2][BATCH][HDIM + 8];
  int tid = threadIdx.x;
  int lane = tid & 31, warp = tid >> 5;
  int j0 = warp * 16;

  for (int idx = tid; idx < BATCH * HDIM; idx += 512) {
    int b = idx >> 8, j = idx & 255;
    h_sh[0][b][j] = (bf16_t)h0[dir * BATCH * HDIM + idx];
  }
  float creg[2][8];
#pragma unroll
  for (int nt = 0; nt < 2; ++nt)
#pragma unroll
    for (int r = 0; r < 8; ++r) {
      int b = nt * 16 + (lane & 15);
      int j = j0 + r + 8 * (lane >> 4);
      creg[nt][r] = c0[dir * BATCH * HDIM + b * HDIM + j];
    }
  __syncthreads();

  int cur = 0;
  for (int s = 0; s < T_SEQ; ++s) {
    int t = dir ? (T_SEQ - 1 - s) : s;

    // Prefetch next step's gate pre-activations (exactly what this wave reads)
    {
      int sn = (s + 1 < T_SEQ) ? (s + 1) : s;
      int tn = dir ? (T_SEQ - 1 - sn) : sn;
      const float* Gn = Gd + (size_t)tn * HID4 * BATCH;
#pragma unroll
      for (int g = 0; g < 4; ++g)
#pragma unroll
        for (int nt = 0; nt < 2; ++nt)
#pragma unroll
          for (int r = 0; r < 8; ++r) {
            int b = nt * 16 + (lane & 15);
            int j = j0 + r + 8 * (lane >> 4);
            __builtin_prefetch(&Gn[(g * HDIM + j) * BATCH + b], 0, 3);
          }
    }

    v8f acc[4][2];
#pragma unroll
    for (int g = 0; g < 4; ++g)
#pragma unroll
      for (int nt = 0; nt < 2; ++nt)
#pragma unroll
        for (int r = 0; r < 8; ++r) acc[g][nt][r] = 0.0f;
#pragma unroll
    for (int kt = 0; kt < 8; ++kt) {
      int k0 = kt * 32;
      v16bf bfr[2];
#pragma unroll
      for (int nt = 0; nt < 2; ++nt) {
        int bcol = nt * 16 + (lane & 15);
        int kb = k0 + (lane >> 4) * 16;
        const bf16_t* base = &h_sh[cur][bcol][kb];
        v8bf lo = *(const v8bf*)base;
        v8bf hi = *(const v8bf*)(base + 8);
        bfr[nt] = cat16(lo, hi);
      }
#pragma unroll
      for (int g = 0; g < 4; ++g) {
        v16bf afr = load_a_frag(Whh, HDIM, g * HDIM + j0, k0, lane);
        acc[g][0] = wmma_bf16(afr, bfr[0], acc[g][0]);
        acc[g][1] = wmma_bf16(afr, bfr[1], acc[g][1]);
      }
    }

    const float* Gt = Gd + (size_t)t * HID4 * BATCH;
#pragma unroll
    for (int nt = 0; nt < 2; ++nt) {
#pragma unroll
      for (int r = 0; r < 8; ++r) {
        int b = nt * 16 + (lane & 15);
        int j = j0 + r + 8 * (lane >> 4);
        float pi = acc[0][nt][r] + Gt[(0 * HDIM + j) * BATCH + b];
        float pf = acc[1][nt][r] + Gt[(1 * HDIM + j) * BATCH + b];
        float pg = acc[2][nt][r] + Gt[(2 * HDIM + j) * BATCH + b];
        float po = acc[3][nt][r] + Gt[(3 * HDIM + j) * BATCH + b];
        float cc = sigmoidf_(pf) * creg[nt][r] + sigmoidf_(pi) * tanhf(pg);
        creg[nt][r] = cc;
        float hv = sigmoidf_(po) * tanhf(cc);
        hout[((size_t)t * BATCH + b) * HDIM + j] = hv;
        h_sh[cur ^ 1][b][j] = (bf16_t)hv;  // write ping-pong buffer
      }
    }
    __syncthreads();  // single barrier per step: new buffer visible
    cur ^= 1;
  }
}

// -------- feats = [hf,hr] @ W_lin^T + b_lin : one wave per (t,b) row --------
__global__ __launch_bounds__(256) void feats_kernel(const float* __restrict__ hf,
                                                    const float* __restrict__ hr,
                                                    const float* __restrict__ Wl,
                                                    const float* __restrict__ bl,
                                                    float* __restrict__ feats) {
  int lane = threadIdx.x & 31, warp = threadIdx.x >> 5;
  int row = blockIdx.x * 8 + warp;  // 0..16383 = t*32+b
  const float* hfr = hf + (size_t)row * HDIM;
  const float* hrr = hr + (size_t)row * HDIM;
  for (int k = 0; k < KTAGS; ++k) {
    float s = 0.0f;
#pragma unroll
    for (int jj = 0; jj < 8; ++jj) {
      int j = jj * 32 + lane;
      s += hfr[j] * Wl[k * 512 + j] + hrr[j] * Wl[k * 512 + 256 + j];
    }
    for (int off = 16; off > 0; off >>= 1) s += __shfl_down(s, off);
    if (lane == 0) feats[row * KTAGS + k] = s + bl[k];
  }
}

// -------- CRF: gold score + forward DP, one lane per batch --------
__global__ __launch_bounds__(32) void crf_kernel(const float* __restrict__ feats,
                                                 const int* __restrict__ tags,
                                                 const float* __restrict__ st,
                                                 const float* __restrict__ en,
                                                 const float* __restrict__ trans,
                                                 float* __restrict__ out) {
  __shared__ float tr_s[KTAGS * KTAGS], st_s[KTAGS], en_s[KTAGS];
  int lane = threadIdx.x;
  for (int i = lane; i < KTAGS * KTAGS; i += 32) tr_s[i] = trans[i];
  if (lane < KTAGS) { st_s[lane] = st[lane]; en_s[lane] = en[lane]; }
  __syncthreads();

  int b = lane;
  float alpha[KTAGS];
  const float* f0 = feats + b * KTAGS;
  for (int k = 0; k < KTAGS; ++k) alpha[k] = st_s[k] + f0[k];

  int tprev = tags[b * T_SEQ + 0];
  float score = st_s[tprev] + f0[tprev];
  for (int t = 1; t < T_SEQ; ++t) {
    int tg = tags[b * T_SEQ + t];
    score += tr_s[tprev * KTAGS + tg] + feats[((size_t)t * BATCH + b) * KTAGS + tg];
    tprev = tg;
  }
  score += en_s[tprev];

  for (int t = 1; t < T_SEQ; ++t) {
    const float* em = feats + ((size_t)t * BATCH + b) * KTAGS;
    float na[KTAGS];
    for (int k2 = 0; k2 < KTAGS; ++k2) {
      float m = -3.4e38f;
      for (int k1 = 0; k1 < KTAGS; ++k1)
        m = fmaxf(m, alpha[k1] + tr_s[k1 * KTAGS + k2]);
      float ss = 0.0f;
      for (int k1 = 0; k1 < KTAGS; ++k1)
        ss += __expf(alpha[k1] + tr_s[k1 * KTAGS + k2] - m);
      na[k2] = m + __logf(ss) + em[k2];
    }
    for (int k = 0; k < KTAGS; ++k) alpha[k] = na[k];
  }
  float m = -3.4e38f;
  for (int k = 0; k < KTAGS; ++k) m = fmaxf(m, alpha[k] + en_s[k]);
  float ss = 0.0f;
  for (int k = 0; k < KTAGS; ++k) ss += __expf(alpha[k] + en_s[k] - m);
  float val = score - (m + __logf(ss));
  for (int off = 16; off > 0; off >>= 1) val += __shfl_down(val, off);
  if (lane == 0) out[0] = -val / (float)BATCH;
}

extern "C" void kernel_launch(void* const* d_in, const int* in_sizes, int n_in,
                              void* d_out, int out_size, void* d_ws, size_t ws_size,
                              hipStream_t stream) {
  const int*   texts = (const int*)d_in[0];
  const int*   tags  = (const int*)d_in[1];
  const float* emb   = (const float*)d_in[2];
  const float* Wih_f = (const float*)d_in[3];
  const float* Whh_f = (const float*)d_in[4];
  const float* b_f   = (const float*)d_in[5];
  const float* Wih_r = (const float*)d_in[6];
  const float* Whh_r = (const float*)d_in[7];
  const float* b_r   = (const float*)d_in[8];
  const float* W_lin = (const float*)d_in[9];
  const float* b_lin = (const float*)d_in[10];
  const float* st    = (const float*)d_in[11];
  const float* en    = (const float*)d_in[12];
  const float* trans = (const float*)d_in[13];
  const float* h0    = (const float*)d_in[14];
  const float* c0    = (const float*)d_in[15];
  float* out = (float*)d_out;

  char* w = (char*)d_ws;
  auto carve = [&](size_t bytes) -> char* {
    char* p = w;
    w += (bytes + 255) & ~(size_t)255;
    return p;
  };
  bf16_t* xbf  = (bf16_t*)carve((size_t)16384 * 256 * 2);  // 8 MB
  bf16_t* wihf = (bf16_t*)carve((size_t)1024 * 256 * 2);
  bf16_t* whhf = (bf16_t*)carve((size_t)1024 * 256 * 2);
  bf16_t* wihr = (bf16_t*)carve((size_t)1024 * 256 * 2);
  bf16_t* whhr = (bf16_t*)carve((size_t)1024 * 256 * 2);
  float*  Gfb  = (float*)carve((size_t)512 * 1024 * 32 * 4);  // 64 MB
  float*  Grb  = (float*)carve((size_t)512 * 1024 * 32 * 4);  // 64 MB
  float*  hf   = (float*)carve((size_t)16384 * 256 * 4);      // 16 MB
  float*  hr   = (float*)carve((size_t)16384 * 256 * 4);      // 16 MB
  float*  fts  = (float*)carve((size_t)16384 * 9 * 4);

  cvt_kernel<<<1024, 256, 0, stream>>>(Wih_f, wihf, 1024 * 256);
  cvt_kernel<<<1024, 256, 0, stream>>>(Whh_f, whhf, 1024 * 256);
  cvt_kernel<<<1024, 256, 0, stream>>>(Wih_r, wihr, 1024 * 256);
  cvt_kernel<<<1024, 256, 0, stream>>>(Whh_r, whhr, 1024 * 256);
  embed_kernel<<<16384, 256, 0, stream>>>(texts, emb, xbf);
  ip_gemm_kernel<<<2048, 256, 0, stream>>>(xbf, wihf, b_f, Gfb);
  ip_gemm_kernel<<<2048, 256, 0, stream>>>(xbf, wihr, b_r, Grb);
  lstm_kernel<<<2, 512, 0, stream>>>(Gfb, Grb, whhf, whhr, h0, c0, hf, hr);
  feats_kernel<<<2048, 256, 0, stream>>>(hf, hr, W_lin, b_lin, fts);
  crf_kernel<<<1, 32, 0, stream>>>(fts, tags, st, en, trans, out);
  (void)in_sizes; (void)n_in; (void)out_size; (void)ws_size;
}
